// TransformerBlock_27762668601707
// MI455X (gfx1250) — matrix operand order repaired and checked by reading, not compiled, not run on hardware
//
#include <hip/hip_runtime.h>
#include <math.h>

// ---------------------------------------------------------------------------
// CDNA5 (gfx1250) implementation. wave32. All matmuls via
// v_wmma_f32_16x16x32_bf16 (D = A(16x32 bf16) * B(32x16 bf16) + C(16x16 f32)).
//
// Dead code in the reference (verified): epa = x_ca, so the sparse-attention
// branch (index_sample, top_k, ctx scatter, W_out/b_out) never affects the
// output. We skip it entirely.
//
// v2: register-blocked WMMA (each wave computes a 16x64 tile: one A fragment
// reused across 4 B fragments / 4 accumulators) for the two dense GEMMs and
// the implicit-GEMM conv. Cuts A-side global traffic 4x and doubles WMMA
// density per vmem issue.
// ---------------------------------------------------------------------------

typedef __bf16 bf16_t;
typedef __attribute__((ext_vector_type(16))) __bf16 v16bf;
typedef __attribute__((ext_vector_type(8)))  __bf16 v8bf;
typedef __attribute__((ext_vector_type(8)))  float  v8f;

#define WMMA_BF16(a, b, c) \
  __builtin_amdgcn_wmma_f32_16x16x32_bf16(false, (a), false, (b), (short)0, (c), false, false)

// Fragment loader: two contiguous 8-element (16B) bf16 chunks -> v16bf.
// Used with contiguous K for A 16x32 (lanes 0-15: K 0-15;
// lanes 16-31: K 16-31) and the same order for B 32x16 (lanes 0-15: K 0-15,
// lanes 16-31: K 16-31) when fed the right base pointers.
__device__ __forceinline__ v16bf load_frag(const bf16_t* lo, const bf16_t* hi) {
  v8bf l = *(const v8bf*)lo;
  v8bf h = *(const v8bf*)hi;
  return __builtin_shufflevector(l, h, 0,1,2,3,4,5,6,7,8,9,10,11,12,13,14,15);
}

// Problem constants
#define BB 2
#define CC 256
#define NN 4096          // 16*16*16
#define NH 8
#define DH 32
#define BNC (BB*NN*CC)   // 2097152

// ---------------------------------------------------------------------------
// 1) x (B,C,N) f32  ->  xs (B,N,C) f32  + bf16 copy
__global__ void k_transpose_x(const float* __restrict__ x,
                              float* __restrict__ xs, bf16_t* __restrict__ xsb) {
  int idx = blockIdx.x * blockDim.x + threadIdx.x;
  if (idx >= BNC) return;
  int c = idx & 255;
  int n = (idx >> 8) & 4095;
  int b = idx >> 20;
  float v = x[((size_t)(b * CC + c)) * NN + n];
  xs[idx]  = v;
  xsb[idx] = (bf16_t)v;
}

// 2) generic transpose-convert: W (R x Cc) f32 -> WT (Cc x R) bf16
__global__ void k_transpose_w(const float* __restrict__ in, bf16_t* __restrict__ out,
                              int R, int Cc) {
  int idx = blockIdx.x * blockDim.x + threadIdx.x;
  if (idx >= R * Cc) return;
  int r = idx / Cc, c = idx - r * Cc;
  out[(size_t)c * R + r] = (bf16_t)in[idx];
}

// 3) conv_w (Co,Ci,3,3,3) f32 -> wT[tap][co][ci] bf16
__global__ void k_pack_convw(const float* __restrict__ w, bf16_t* __restrict__ out) {
  int idx = blockIdx.x * blockDim.x + threadIdx.x;
  if (idx >= 27 * CC * CC) return;
  int ci = idx & 255;
  int co = (idx >> 8) & 255;
  int tap = idx >> 16;
  out[idx] = (bf16_t)w[((size_t)(co * CC + ci)) * 27 + tap];
}

// ---------------------------------------------------------------------------
// 4) Blocked WMMA GEMM: C(MxN,f32) = A(MxK,bf16 rowmajor) * BT(NxK,bf16)^T
//    One wave computes a 16x64 tile (A frag reused over 4 B frags).
//    Requires N % 64 == 0.
__global__ __launch_bounds__(256)
void k_gemm_bf16(const bf16_t* __restrict__ A, const bf16_t* __restrict__ BT,
                 float* __restrict__ C, int M, int N, int K) {
  int wave = (blockIdx.x * blockDim.x + threadIdx.x) >> 5;
  int lane = threadIdx.x & 31;
  int ng = N >> 6;                                  // 64-col groups
  int tm = wave / ng, tg = wave - tm * ng;
  if (tm * 16 >= M) return;
  int row = lane & 15, half = lane >> 4;
  const bf16_t* a0 = A  + (size_t)(tm * 16 + row) * K;
  const bf16_t* b0 = BT + (size_t)(tg * 64 + row) * K;
  v8f acc0 = {}, acc1 = {}, acc2 = {}, acc3 = {};
  for (int k0 = 0; k0 < K; k0 += 32) {
    v16bf a = load_frag(a0 + k0 + 16 * half, a0 + k0 + 16 * half + 8);
    v16bf bb0 = load_frag(b0 + k0 + 16 * half,              b0 + k0 + 16 * half + 8);
    v16bf bb1 = load_frag(b0 + 16 * K + k0 + 16 * half,     b0 + 16 * K + k0 + 16 * half + 8);
    v16bf bb2 = load_frag(b0 + 32 * K + k0 + 16 * half,     b0 + 32 * K + k0 + 16 * half + 8);
    v16bf bb3 = load_frag(b0 + 48 * K + k0 + 16 * half,     b0 + 48 * K + k0 + 16 * half + 8);
    acc0 = WMMA_BF16(a, bb0, acc0);
    acc1 = WMMA_BF16(a, bb1, acc1);
    acc2 = WMMA_BF16(a, bb2, acc2);
    acc3 = WMMA_BF16(a, bb3, acc3);
  }
  float* c0 = C + (size_t)(tm * 16 + 8 * half) * N + tg * 64 + row;
#pragma unroll
  for (int r = 0; r < 8; ++r) {
    c0[(size_t)r * N]      = acc0[r];
    c0[(size_t)r * N + 16] = acc1[r];
    c0[(size_t)r * N + 32] = acc2[r];
    c0[(size_t)r * N + 48] = acc3[r];
  }
}

// 5) Blocked WMMA GEMM + fused epilogue for out2 (M=8192, N=K=256):
//    attn = xs + gamma1[c] * (xca @ W_out2 + b_out2[c]); also bf16 copy.
__global__ __launch_bounds__(256)
void k_gemm_out2(const bf16_t* __restrict__ A, const bf16_t* __restrict__ BT,
                 const float* __restrict__ xs, const float* __restrict__ bias,
                 const float* __restrict__ gamma,
                 float* __restrict__ attnf, bf16_t* __restrict__ attnb) {
  const int M = BB * NN, N = CC, K = CC;
  int wave = (blockIdx.x * blockDim.x + threadIdx.x) >> 5;
  int lane = threadIdx.x & 31;
  int ng = N >> 6;
  int tm = wave / ng, tg = wave - tm * ng;
  if (tm * 16 >= M) return;
  int row = lane & 15, half = lane >> 4;
  const bf16_t* a0 = A  + (size_t)(tm * 16 + row) * K;
  const bf16_t* b0 = BT + (size_t)(tg * 64 + row) * K;
  v8f acc0 = {}, acc1 = {}, acc2 = {}, acc3 = {};
  for (int k0 = 0; k0 < K; k0 += 32) {
    v16bf a = load_frag(a0 + k0 + 16 * half, a0 + k0 + 16 * half + 8);
    v16bf bb0 = load_frag(b0 + k0 + 16 * half,          b0 + k0 + 16 * half + 8);
    v16bf bb1 = load_frag(b0 + 16 * K + k0 + 16 * half, b0 + 16 * K + k0 + 16 * half + 8);
    v16bf bb2 = load_frag(b0 + 32 * K + k0 + 16 * half, b0 + 32 * K + k0 + 16 * half + 8);
    v16bf bb3 = load_frag(b0 + 48 * K + k0 + 16 * half, b0 + 48 * K + k0 + 16 * half + 8);
    acc0 = WMMA_BF16(a, bb0, acc0);
    acc1 = WMMA_BF16(a, bb1, acc1);
    acc2 = WMMA_BF16(a, bb2, acc2);
    acc3 = WMMA_BF16(a, bb3, acc3);
  }
  v8f accs[4] = {acc0, acc1, acc2, acc3};
#pragma unroll
  for (int j = 0; j < 4; ++j) {
    int c = tg * 64 + j * 16 + row;
    float g = gamma[c], bo = bias[c];
#pragma unroll
    for (int r = 0; r < 8; ++r) {
      size_t o = (size_t)(tm * 16 + 8 * half + r) * N + c;
      float v = xs[o] + g * (accs[j][r] + bo);
      attnf[o] = v;
      attnb[o] = (bf16_t)v;
    }
  }
}

// ---------------------------------------------------------------------------
// 6) L2-norm along N per (b, channel) for q,k; head-major bf16 repack.
//    qkv layout: (B, N, 768) f32 with cols 0..255=q, 256..511=k, 512..767=v_ca
//    qT/kT: (B, nh, dh, N) bf16 (for contiguous-K WMMA A/B frags)
//    vb   : (B, N, C) bf16
__global__ __launch_bounds__(256)
void k_l2norm_pack(const float* __restrict__ qkv, bf16_t* __restrict__ qT,
                   bf16_t* __restrict__ kT, bf16_t* __restrict__ vb) {
  int b = blockIdx.x / 768;
  int c = blockIdx.x - b * 768;
  int tid = threadIdx.x;
  const float* base = qkv + (size_t)b * NN * 768 + c;
  __shared__ float red[256];
  __shared__ float sInv;
  float inv = 1.f;
  if (c < 512) {
    float s = 0.f;
    for (int n = tid; n < NN; n += 256) { float v = base[(size_t)n * 768]; s += v * v; }
    red[tid] = s; __syncthreads();
    for (int st = 128; st > 0; st >>= 1) {
      if (tid < st) red[tid] += red[tid + st];
      __syncthreads();
    }
    if (tid == 0) sInv = 1.f / fmaxf(sqrtf(red[0]), 1e-12f);
    __syncthreads();
    inv = sInv;
  }
  if (c < 256) {
    bf16_t* o = qT + ((size_t)(b * NH + (c >> 5)) * DH + (c & 31)) * NN;
    for (int n = tid; n < NN; n += 256) o[n] = (bf16_t)(base[(size_t)n * 768] * inv);
  } else if (c < 512) {
    int cc = c - 256;
    bf16_t* o = kT + ((size_t)(b * NH + (cc >> 5)) * DH + (cc & 31)) * NN;
    for (int n = tid; n < NN; n += 256) o[n] = (bf16_t)(base[(size_t)n * 768] * inv);
  } else {
    bf16_t* o = vb + (size_t)b * NN * CC + (c - 512);
    for (int n = tid; n < NN; n += 256) o[(size_t)n * CC] = (bf16_t)(base[(size_t)n * 768]);
  }
}

// ---------------------------------------------------------------------------
// 7) Cross-attention per (b,h): S = qT*kT^T (32x32 over K=4096), softmax rows,
//    x_ca = A * v^T (32x4096). One block = 4 waves per (b,h).
__global__ __launch_bounds__(128)
void k_attn_ca(const bf16_t* __restrict__ qT, const bf16_t* __restrict__ kT,
               const bf16_t* __restrict__ vb, bf16_t* __restrict__ xca) {
  int b = blockIdx.x >> 3, h = blockIdx.x & 7;
  int w = threadIdx.x >> 5, lane = threadIdx.x & 31;
  int row = lane & 15, half = lane >> 4;
  __shared__ float  S[32][32];
  __shared__ bf16_t At[32][32];
  // Phase 1: S[d][e] = sum_n q[d][n]*k[e][n]; wave w -> tile (d:w&1, e:w>>1)
  {
    int dt = w & 1, et = w >> 1;
    const bf16_t* a0 = qT + ((size_t)(b * NH + h) * DH + dt * 16 + row) * NN;
    const bf16_t* b0 = kT + ((size_t)(b * NH + h) * DH + et * 16 + row) * NN;
    v8f acc = {};
    for (int k0 = 0; k0 < NN; k0 += 32) {
      v16bf a  = load_frag(a0 + k0 + 16 * half, a0 + k0 + 16 * half + 8);
      v16bf bb = load_frag(b0 + k0 + 16 * half, b0 + k0 + 16 * half + 8);
      acc = WMMA_BF16(a, bb, acc);
    }
#pragma unroll
    for (int r = 0; r < 8; ++r) S[dt * 16 + 8 * half + r][et * 16 + row] = acc[r];
  }
  __syncthreads();
  // Phase 2: row softmax (32 rows, 32 cols)
  if (threadIdx.x < 32) {
    int r = threadIdx.x;
    float m = -1e30f;
    for (int j = 0; j < 32; ++j) m = fmaxf(m, S[r][j]);
    float s = 0.f;
    for (int j = 0; j < 32; ++j) s += __expf(S[r][j] - m);
    float invs = 1.f / s;
    for (int j = 0; j < 32; ++j) At[r][j] = (bf16_t)(__expf(S[r][j] - m) * invs);
  }
  __syncthreads();
  // Phase 3: x_ca[d][n] = sum_e At[d][e] * v[n][e]   (K=32, one WMMA per tile)
  int dt = w & 1;
  v16bf a;
#pragma unroll
  for (int i = 0; i < 8; ++i) {
    a[i]     = At[dt * 16 + row][16 * half + i];
    a[8 + i] = At[dt * 16 + row][16 * half + 8 + i];
  }
  for (int nt = (w >> 1); nt < 256; nt += 2) {
    const bf16_t* b0 = vb + ((size_t)(b * NN + nt * 16 + row)) * CC + h * DH + 16 * half;
    v16bf bb = load_frag(b0, b0 + 8);
    v8f acc = {};
    acc = WMMA_BF16(a, bb, acc);
    v8bf o;
#pragma unroll
    for (int r = 0; r < 8; ++r) o[r] = (bf16_t)acc[r];
    *(v8bf*)(xca + ((size_t)(b * NN + nt * 16 + row)) * CC + h * DH + dt * 16 + 8 * half) = o;
  }
}

// ---------------------------------------------------------------------------
// 8) 3x3x3 conv, pad 1, as blocked implicit WMMA GEMM.
//    A rows = 16 consecutive output positions (fixed z,y; x=0..15); one A frag
//    per (tap, ci-chunk) reused over 4 co-tiles (64 output channels per wave).
//    z/y tap culling is wave-uniform; x padding masks only the A loads
//    (EXEC stays all-1 at the WMMA).
__global__ __launch_bounds__(256)
void k_conv3d(const bf16_t* __restrict__ attnb, const bf16_t* __restrict__ wT,
              float* __restrict__ y) {
  int wave = (blockIdx.x * blockDim.x + threadIdx.x) >> 5;
  int lane = threadIdx.x & 31;
  int tg = wave & 3, tm = wave >> 2;           // 512 row-tiles x 4 co-groups(64)
  if (tm >= BB * 256) return;
  int b  = tm >> 8;
  int p0 = (tm & 255) * 16;                    // p = z*256 + y*16 + x
  int z  = p0 >> 8, y0 = (p0 >> 4) & 15;
  int row = lane & 15, half = lane >> 4;
  v8f acc0 = {}, acc1 = {}, acc2 = {}, acc3 = {};
  for (int tap = 0; tap < 27; ++tap) {
    int dz = tap / 9 - 1, dy = (tap / 3) % 3 - 1, dx = tap % 3 - 1;
    int zz = z + dz, yy = y0 + dy;
    if ((unsigned)zz >= 16u || (unsigned)yy >= 16u) continue;   // uniform
    int xx = row + dx;
    bool xok = (unsigned)xx < 16u;                              // per-lane
    const bf16_t* a0 = attnb + ((size_t)(b * NN + zz * 256 + yy * 16 + xx)) * CC;
    const bf16_t* w0 = wT + ((size_t)(tap * CC + tg * 64 + row)) * CC;
    for (int c0 = 0; c0 < CC; c0 += 32) {
      v16bf a = {};
      if (xok) a = load_frag(a0 + c0 + 16 * half, a0 + c0 + 16 * half + 8);
      v16bf bb0 = load_frag(w0 + c0 + 16 * half,            w0 + c0 + 16 * half + 8);
      v16bf bb1 = load_frag(w0 + 16 * CC + c0 + 16 * half,  w0 + 16 * CC + c0 + 16 * half + 8);
      v16bf bb2 = load_frag(w0 + 32 * CC + c0 + 16 * half,  w0 + 32 * CC + c0 + 16 * half + 8);
      v16bf bb3 = load_frag(w0 + 48 * CC + c0 + 16 * half,  w0 + 48 * CC + c0 + 16 * half + 8);
      acc0 = WMMA_BF16(a, bb0, acc0);
      acc1 = WMMA_BF16(a, bb1, acc1);
      acc2 = WMMA_BF16(a, bb2, acc2);
      acc3 = WMMA_BF16(a, bb3, acc3);
    }
  }
  v8f accs[4] = {acc0, acc1, acc2, acc3};
#pragma unroll
  for (int j = 0; j < 4; ++j)
#pragma unroll
    for (int r = 0; r < 8; ++r)
      y[((size_t)(b * NN + p0 + 8 * half + r)) * CC + tg * 64 + j * 16 + row] = accs[j][r];
}

// ---------------------------------------------------------------------------
// 9) Deterministic two-stage mean/var over each batch (C*N = 1M elems)
__global__ __launch_bounds__(256)
void k_reduce1(const float* __restrict__ y, float* __restrict__ part) {
  int b = blockIdx.x >> 7, blk = blockIdx.x & 127;       // 128 blocks per batch
  const float* base = y + (size_t)b * (NN * CC) + (size_t)blk * 8192;
  float s = 0.f, s2 = 0.f;
  for (int i = threadIdx.x; i < 8192; i += 256) { float v = base[i]; s += v; s2 += v * v; }
  __shared__ float rs[256], rq[256];
  rs[threadIdx.x] = s; rq[threadIdx.x] = s2; __syncthreads();
  for (int st = 128; st > 0; st >>= 1) {
    if (threadIdx.x < st) { rs[threadIdx.x] += rs[threadIdx.x + st]; rq[threadIdx.x] += rq[threadIdx.x + st]; }
    __syncthreads();
  }
  if (threadIdx.x == 0) { part[blockIdx.x * 2] = rs[0]; part[blockIdx.x * 2 + 1] = rq[0]; }
}

__global__ __launch_bounds__(128)
void k_reduce2(const float* __restrict__ part, float* __restrict__ stats) {
  int b = blockIdx.x, tid = threadIdx.x;
  __shared__ float rs[128], rq[128];
  rs[tid] = part[(b * 128 + tid) * 2];
  rq[tid] = part[(b * 128 + tid) * 2 + 1];
  __syncthreads();
  for (int st = 64; st > 0; st >>= 1) {
    if (tid < st) { rs[tid] += rs[tid + st]; rq[tid] += rq[tid + st]; }
    __syncthreads();
  }
  if (tid == 0) {
    const float inv = 1.f / (float)(NN * CC);
    float mu  = rs[0] * inv;
    float var = rq[0] * inv - mu * mu;
    stats[b * 2]     = mu;
    stats[b * 2 + 1] = rsqrtf(var + 1e-5f);
  }
}

// 10) normalize + affine + residual + leaky-ReLU; output layout (B, C, N)
__global__ __launch_bounds__(256)
void k_final(const float* __restrict__ y, const float* __restrict__ attnf,
             const float* __restrict__ stats, const float* __restrict__ gnw,
             const float* __restrict__ gnb, float* __restrict__ out) {
  int idx = blockIdx.x * blockDim.x + threadIdx.x;
  if (idx >= BNC) return;
  int n = idx & 4095;
  int c = (idx >> 12) & 255;
  int b = idx >> 20;
  size_t src = ((size_t)(b * NN + n)) * CC + c;
  float v = (y[src] - stats[b * 2]) * stats[b * 2 + 1] * gnw[c] + gnb[c] + attnf[src];
  out[idx] = v >= 0.f ? v : 0.01f * v;
}

// ---------------------------------------------------------------------------
extern "C" void kernel_launch(void* const* d_in, const int* in_sizes, int n_in,
                              void* d_out, int out_size, void* d_ws, size_t ws_size,
                              hipStream_t stream) {
  const float* x      = (const float*)d_in[0];
  const float* W_qkvv = (const float*)d_in[1];
  // d_in[2]=W_out, d_in[3]=b_out, d_in[10]=index_sample: dead code in reference
  const float* W_out2 = (const float*)d_in[4];
  const float* b_out2 = (const float*)d_in[5];
  const float* gamma1 = (const float*)d_in[6];
  const float* conv_w = (const float*)d_in[7];
  const float* gn_w   = (const float*)d_in[8];
  const float* gn_b   = (const float*)d_in[9];
  float* out = (float*)d_out;

  char* w = (char*)d_ws;
  size_t off = 0;
  auto alloc = [&](size_t bytes) -> char* {
    char* p = w + off;
    off += (bytes + 255) & ~(size_t)255;
    return p;
  };
  float*  xs_f    = (float*) alloc((size_t)BNC * 4);
  bf16_t* xs_b    = (bf16_t*)alloc((size_t)BNC * 2);
  float*  qkv_f   = (float*) alloc((size_t)BB * NN * 768 * 4);
  bf16_t* WqkvT   = (bf16_t*)alloc((size_t)1024 * 256 * 2);
  bf16_t* qT      = (bf16_t*)alloc((size_t)BNC * 2);
  bf16_t* kT      = (bf16_t*)alloc((size_t)BNC * 2);
  bf16_t* v_b     = (bf16_t*)alloc((size_t)BNC * 2);
  bf16_t* xca_b   = (bf16_t*)alloc((size_t)BNC * 2);
  bf16_t* Wout2T  = (bf16_t*)alloc((size_t)256 * 256 * 2);
  float*  attn_f  = (float*) alloc((size_t)BNC * 4);
  bf16_t* attn_b  = (bf16_t*)alloc((size_t)BNC * 2);
  bf16_t* wconvT  = (bf16_t*)alloc((size_t)27 * 256 * 256 * 2);
  float*  y_f     = (float*) alloc((size_t)BNC * 4);
  float*  part    = (float*) alloc((size_t)256 * 2 * 4);
  float*  stats   = (float*) alloc((size_t)4 * 4);
  (void)ws_size; (void)in_sizes; (void)n_in; (void)out_size;

  // Stage inputs
  k_transpose_x<<<BNC / 256, 256, 0, stream>>>(x, xs_f, xs_b);
  k_transpose_w<<<(256 * 1024) / 256, 256, 0, stream>>>(W_qkvv, WqkvT, 256, 1024);
  k_transpose_w<<<(256 * 256) / 256, 256, 0, stream>>>(W_out2, Wout2T, 256, 256);
  k_pack_convw<<<(27 * 256 * 256) / 256, 256, 0, stream>>>(conv_w, wconvT);

  // qkv GEMM: (8192 x 256) * (256 x 768) -> only q,k,v_ca columns
  {
    int waves = (BB * NN / 16) * (768 / 64);   // 6144 (16x64 tiles per wave)
    k_gemm_bf16<<<waves / 8, 256, 0, stream>>>(xs_b, WqkvT, qkv_f, BB * NN, 768, 256);
  }

  // L2-norm along N + head-major repack
  k_l2norm_pack<<<BB * 768, 256, 0, stream>>>(qkv_f, qT, kT, v_b);

  // Cross-attention per (b,h)
  k_attn_ca<<<BB * NH, 128, 0, stream>>>(qT, kT, v_b, xca_b);

  // out2 GEMM + residual/gamma epilogue -> attn (f32 + bf16)
  {
    int waves = (BB * NN / 16) * (256 / 64);   // 2048
    k_gemm_out2<<<waves / 8, 256, 0, stream>>>(xca_b, Wout2T, xs_f, b_out2, gamma1,
                                               attn_f, attn_b);
  }

  // 3x3x3 conv (blocked implicit WMMA GEMM)
  {
    int waves = (BB * 256) * 4;                // 2048
    k_conv3d<<<waves / 8, 256, 0, stream>>>(attn_b, wconvT, y_f);
  }

  // per-batch mean/var, then fused epilogue
  k_reduce1<<<BB * 128, 256, 0, stream>>>(y_f, part);
  k_reduce2<<<BB, 128, 0, stream>>>(part, stats);
  k_final<<<BNC / 256, 256, 0, stream>>>(y_f, attn_f, stats, gn_w, gn_b, out);
}
